// NetTransform_2731599200310
// MI455X (gfx1250) — matrix operand order repaired and checked by reading, not compiled, not run on hardware
//
#include <hip/hip_runtime.h>
#include <hip/hip_bf16.h>

typedef __bf16 bf16_t;
typedef __attribute__((ext_vector_type(16))) __bf16 v16bf;
typedef __attribute__((ext_vector_type(8)))  float  v8f;
typedef __attribute__((ext_vector_type(4)))  unsigned int u32x4;
typedef __attribute__((ext_vector_type(4)))  int i32x4;
typedef __attribute__((ext_vector_type(8)))  int i32x8;

#define BSZ 16
#define TSZ 512
#define IDIM 80
#define HDIM 512
#define NROW (BSZ * TSZ)
#define IPAD 96

__device__ __forceinline__ v8f wmma_bf16(v16bf a, v16bf b, v8f c) {
  return __builtin_amdgcn_wmma_f32_16x16x32_bf16(false, a, false, b, (short)0, c,
                                                 false, false);
}

__device__ __forceinline__ float wave_reduce(float v) {
#pragma unroll
  for (int off = 16; off > 0; off >>= 1) v += __shfl_xor(v, off, 32);
  return v;
}

// CDNA5 async Global->LDS copy (ASYNCcnt-tracked), 16B per lane.
__device__ __forceinline__ void async_copy16(unsigned ldsaddr, const void* g) {
  asm volatile("global_load_async_to_lds_b128 %0, %1, off" ::"v"(ldsaddr),
               "v"((unsigned long long)g)
               : "memory");
}
__device__ __forceinline__ void async_wait0() {
  asm volatile("s_wait_asynccnt 0" ::: "memory");
}
__device__ __forceinline__ unsigned lds_addr(const void* p) {
  return (unsigned)(size_t)p;  // flat shared aperture: low 32 bits = LDS offset
}

// ---------------------------------------------------------------- prep
__global__ void init_kernel(float* acc) {
  if (threadIdx.x < 2) acc[threadIdx.x] = 0.0f;
}

__global__ void combine_kernel(const float* acc, float* out) {
  out[0] = acc[0] + acc[1];
}

__global__ void cast_pad_kernel(const float* __restrict__ src,
                                bf16_t* __restrict__ dst, int srows, int scols,
                                int drows, int dcols) {
  int idx = blockIdx.x * blockDim.x + threadIdx.x;
  int total = drows * dcols;
  if (idx >= total) return;
  int r = idx / dcols, c = idx % dcols;
  float v = (r < srows && c < scols) ? src[r * scols + c] : 0.0f;
  dst[idx] = (bf16_t)v;
}

// Fragment-major swizzles: one thread emits one lane's 16 contiguous bf16.
// B layout: element e <-> K = e + 16*half, N = lane&15
__global__ void swizzle_b_kernel(const float* __restrict__ src,
                                 bf16_t* __restrict__ dst, int K, int N, int KT,
                                 int NT) {
  int t = blockIdx.x * blockDim.x + threadIdx.x;
  if (t >= KT * NT * 32) return;
  int lane = t & 31, tile = t >> 5;
  int nt = tile % NT, kc = tile / NT;
  int half = lane >> 4, nn = lane & 15;
  int c = nt * 16 + nn;
#pragma unroll
  for (int e = 0; e < 16; ++e) {
    int k = kc * 32 + e + 16 * half;
    float v = (k < K && c < N) ? src[(size_t)k * N + c] : 0.0f;
    dst[(size_t)t * 16 + e] = (bf16_t)v;
  }
}

// A layout: element e <-> K = e + 16*half, M = lane&15
__global__ void swizzle_a_kernel(const float* __restrict__ src,
                                 bf16_t* __restrict__ dst, int M, int K, int KT,
                                 int MT) {
  int t = blockIdx.x * blockDim.x + threadIdx.x;
  if (t >= KT * MT * 32) return;
  int lane = t & 31, tile = t >> 5;
  int mt = tile % MT, kc = tile / MT;
  int half = lane >> 4;
  int m = mt * 16 + (lane & 15);
#pragma unroll
  for (int e = 0; e < 16; ++e) {
    int k = kc * 32 + e + 16 * half;
    float v = (m < M && k < K) ? src[(size_t)m * K + k] : 0.0f;
    dst[(size_t)t * 16 + e] = (bf16_t)v;
  }
}

// ------------------------------------------- GEMM: [NROW,96] x [96,512]
// tmode=1: a = x@Wt1+bt1 -> g1=relu(a), g2=(a-bt1)*r, rm16 packed mask
// tmode=0: h1 = relu(x@We1+be1)
__global__ __launch_bounds__(128) void mlp_in_kernel(
    const bf16_t* __restrict__ xpad, const v16bf* __restrict__ Bswz,
    const float* __restrict__ bias, bf16_t* __restrict__ out1,
    bf16_t* __restrict__ out2, unsigned short* __restrict__ rm16, int tmode) {
  __shared__ __align__(16) bf16_t ldsA[16 * 32];
  int wid = threadIdx.x >> 5, lane = threadIdx.x & 31;
  int half = lane >> 4, nn = lane & 15;
  int row0 = blockIdx.y * 16;
  int col0 = blockIdx.x * 64 + wid * 16;
  int nt = blockIdx.x * 4 + wid;  // N-tile (of 32)
  v8f acc = {};
  for (int kc = 0; kc < 3; ++kc) {
    int k0 = kc * 32;
    __syncthreads();
    if (threadIdx.x < 64) {  // wave-uniform (waves 0,1)
      int r = threadIdx.x >> 2, q = threadIdx.x & 3;
      async_copy16(lds_addr(ldsA) + threadIdx.x * 16u,
                   &xpad[(size_t)(row0 + r) * IPAD + k0 + q * 8]);
    }
    async_wait0();
    __syncthreads();
    v16bf a;
#pragma unroll
    for (int e = 0; e < 16; ++e)
      a[e] = ldsA[(lane & 15) * 32 + e + 16 * half];
    v16bf b = Bswz[(kc * 32 + nt) * 32 + lane];
    acc = wmma_bf16(a, b, acc);
  }
#pragma unroll
  for (int v = 0; v < 8; ++v) {
    int row = row0 + v + 8 * half;
    int col = col0 + nn;
    float aval = acc[v] + bias[col];
    float g = aval > 0.0f ? aval : 0.0f;
    out1[(size_t)row * HDIM + col] = (bf16_t)g;
    if (tmode) {
      out2[(size_t)row * HDIM + col] =
          (bf16_t)(aval > 0.0f ? (aval - bias[col]) : 0.0f);
      unsigned bal = (unsigned)__ballot(aval > 0.0f);
      if (lane == 0) {
        rm16[(size_t)(row0 + v) * 32 + nt] = (unsigned short)(bal & 0xFFFFu);
        rm16[(size_t)(row0 + v + 8) * 32 + nt] = (unsigned short)(bal >> 16);
      }
    }
  }
}

// ------------------------ out_hat = g1@Wt2+bt2 ; e_end = g2@Wt2 ; loss_g
__global__ __launch_bounds__(192) void mlp_t2_kernel(
    const bf16_t* __restrict__ g1, const bf16_t* __restrict__ g2,
    const v16bf* __restrict__ Wt2swz, const float* __restrict__ bt2,
    const float* __restrict__ x, const float* __restrict__ xo,
    const int* __restrict__ ilens, signed char* __restrict__ esign,
    float* __restrict__ accG) {
  __shared__ __align__(16) bf16_t ldsA1[16 * 32];
  __shared__ __align__(16) bf16_t ldsA2[16 * 32];
  int wid = threadIdx.x >> 5, lane = threadIdx.x & 31;
  int half = lane >> 4, nn = lane & 15;
  int row0 = blockIdx.x * 16;
  int col0 = wid * 16;
  v8f acc1 = {}, acc2 = {};
  for (int kc = 0; kc < 16; ++kc) {
    int k0 = kc * 32;
    __syncthreads();
    if (threadIdx.x < 128) {  // wave-uniform (waves 0..3)
      int buf = threadIdx.x >> 6, r = (threadIdx.x >> 2) & 15,
          q = threadIdx.x & 3;
      const bf16_t* s = buf ? g2 : g1;
      unsigned dbase = lds_addr(buf ? ldsA2 : ldsA1);
      async_copy16(dbase + (unsigned)(r * 4 + q) * 16u,
                   &s[(size_t)(row0 + r) * HDIM + k0 + q * 8]);
    }
    async_wait0();
    __syncthreads();
    v16bf a1, a2;
#pragma unroll
    for (int e = 0; e < 16; ++e) {
      int ka = (lane & 15) * 32 + e + 16 * half;
      a1[e] = ldsA1[ka];
      a2[e] = ldsA2[ka];
    }
    v16bf b = Wt2swz[(kc * 6 + wid) * 32 + lane];
    acc1 = wmma_bf16(a1, b, acc1);
    acc2 = wmma_bf16(a2, b, acc2);
  }
  float lsum = 0.0f;
#pragma unroll
  for (int v = 0; v < 8; ++v) {
    int row = row0 + v + 8 * half;
    int col = col0 + nn;
    if (col < IDIM) {
      float outh = acc1[v] + bt2[col];
      float ev = acc2[v];
      esign[(size_t)row * IDIM + col] =
          (signed char)((ev > 0.0f) - (ev < 0.0f));
      int bb = row / TSZ, tt = row % TSZ;
      if (tt < ilens[bb]) {
        float d = outh + x[(size_t)row * IDIM + col] - xo[(size_t)row * IDIM + col];
        lsum += d * d;
      }
    }
  }
  lsum = wave_reduce(lsum);
  if (lane == 0) atomicAdd(accG, lsum);
}

// ------------- heavy kernel: W_n = Wt1 diag(r_n) Wt2 -> raw energy[n]
// No LDS in K-loop: A = swizzled Wt1 fragments (L2-resident), B = swizzled
// Wt2 fragments masked in registers via packed ReLU-mask bits.
__global__ __launch_bounds__(160) void energy_kernel(
    const v16bf* __restrict__ Aswz, const v16bf* __restrict__ Bswz,
    const unsigned short* __restrict__ rm16,
    const signed char* __restrict__ esign, const float* __restrict__ x,
    float* __restrict__ eraw) {
  __shared__ float absx[IDIM], sgnx[IDIM], esg[IDIM];
  __shared__ float redp[5], redn[5];
  int n = blockIdx.x;
  int wid = threadIdx.x >> 5, lane = threadIdx.x & 31;
  int half = lane >> 4, nn = lane & 15;
  if (threadIdx.x < IDIM) {
    float xv = x[(size_t)n * IDIM + threadIdx.x];
    absx[threadIdx.x] = fabsf(xv);
    sgnx[threadIdx.x] = (float)((xv > 0.0f) - (xv < 0.0f));
    esg[threadIdx.x] = (float)esign[(size_t)n * IDIM + threadIdx.x];
  }
  v8f acc[5] = {};
  for (int kc = 0; kc < 16; ++kc) {
    v16bf b = Bswz[(kc * 6 + wid) * 32 + lane];
    unsigned bits = rm16[(size_t)n * 32 + kc * 2 + half];
#pragma unroll
    for (int e = 0; e < 16; ++e)
      if (!((bits >> e) & 1u)) b[e] = (bf16_t)0.0f;
#pragma unroll
    for (int mt = 0; mt < 5; ++mt) {
      v16bf a = Aswz[(kc * 5 + mt) * 32 + lane];
      acc[mt] = wmma_bf16(a, b, acc[mt]);
    }
  }
  __syncthreads();
  float p = 0.0f, q = 0.0f;
#pragma unroll
  for (int mt = 0; mt < 5; ++mt) {
#pragma unroll
    for (int v = 0; v < 8; ++v) {
      int d = mt * 16 + v + 8 * half;
      int k = wid * 16 + nn;
      float wx = acc[mt][v] * sgnx[d] * esg[k];
      p += absx[d] * fmaxf(wx, 0.0f);
      q += absx[d] * fmaxf(-wx, 0.0f);
    }
  }
  p = wave_reduce(p);
  q = wave_reduce(q);
  if (lane == 0) { redp[wid] = p; redn[wid] = q; }
  __syncthreads();
  if (threadIdx.x == 0) {
    float pos = 0.0f, neg = 0.0f;
    for (int w = 0; w < 5; ++w) { pos += redp[w]; neg += redn[w]; }
    float e = neg / pos;  // 1/IDIM factors cancel in the ratio
    if (e != e) e = 1.0f;
    float om = 1.0f - e;
    eraw[n] = 1.0f - om * om;
  }
}

// ------------------------------------------------ global min/max reduce
__global__ void minmax_kernel(const float* __restrict__ eraw,
                              float* __restrict__ mm) {
  __shared__ float smn[256], smx[256];
  float mn = 3.4e38f, mx = -3.4e38f;
  for (int i = threadIdx.x; i < NROW; i += 256) {
    float v = eraw[i];
    mn = fminf(mn, v);
    mx = fmaxf(mx, v);
  }
  smn[threadIdx.x] = mn;
  smx[threadIdx.x] = mx;
  __syncthreads();
  for (int s = 128; s > 0; s >>= 1) {
    if (threadIdx.x < s) {
      smn[threadIdx.x] = fminf(smn[threadIdx.x], smn[threadIdx.x + s]);
      smx[threadIdx.x] = fmaxf(smx[threadIdx.x], smx[threadIdx.x + s]);
    }
    __syncthreads();
  }
  if (threadIdx.x == 0) { mm[0] = smn[0]; mm[1] = smx[0]; }
}

// -------------- per-batch: normalize, rank-threshold, cumprod prefixes
__global__ __launch_bounds__(512) void rank_kernel(
    const float* __restrict__ eraw, const float* __restrict__ mm,
    int* __restrict__ Zpre, float* __restrict__ Lpre) {
  __shared__ float en[TSZ], ef[TSZ], eo[TSZ];
  __shared__ int nneg;
  int b = blockIdx.x, t = threadIdx.x;
  float mn = mm[0], mx = mm[1];
  float e = (eraw[b * TSZ + t] - mn) / (mx - mn);
  ef[t] = e;
  en[t] = 1.0f - e;
  __syncthreads();
  if (t == 0) {
    float s = 0.0f;
    for (int i = 0; i < TSZ; ++i) s += en[i];
    nneg = (int)floorf(s);
  }
  __syncthreads();
  float ei = en[t];
  int rank = 0;
  for (int j = 0; j < TSZ; ++j) {
    float ej = en[j];
    rank += (ej > ei) || (ej == ei && j < t);  // stable descending rank
  }
  eo[t] = (rank < nneg) ? ef[t] : 1.0f;
  __syncthreads();
  if (t == 0) {
    int z = 0;
    float ls = 0.0f;
    for (int i = 0; i < TSZ; ++i) {
      float v = eo[i];
      if (v <= 0.0f) z++; else ls += logf(v);
      Zpre[b * TSZ + i] = z;
      Lpre[b * TSZ + i] = ls;
    }
  }
}

// ------------------------------------ h = h1 @ We2 + be2 ([512x512] K)
// A chunk staged via the Tensor Data Mover when available.
__global__ __launch_bounds__(128) void mlp_h2_kernel(
    const bf16_t* __restrict__ h1, const v16bf* __restrict__ We2swz,
    const float* __restrict__ be2, bf16_t* __restrict__ hout) {
  __shared__ __align__(16) bf16_t ldsA[16 * 32];
  int wid = threadIdx.x >> 5, lane = threadIdx.x & 31;
  int half = lane >> 4, nn = lane & 15;
  int row0 = blockIdx.y * 16;
  int col0 = blockIdx.x * 64 + wid * 16;
  int nt = blockIdx.x * 4 + wid;
  v8f acc = {};
  for (int kc = 0; kc < 16; ++kc) {
    int k0 = kc * 32;
    __syncthreads();
#if __has_builtin(__builtin_amdgcn_tensor_load_to_lds) && \
    __has_builtin(__builtin_amdgcn_s_wait_tensorcnt)
    if (threadIdx.x < 32) {
      // D# group0: count=1, lds_addr, 57-bit global byte addr, type=2
      unsigned long long ga =
          (unsigned long long)(const void*)&h1[(size_t)row0 * HDIM + k0];
      unsigned lds = lds_addr(ldsA);
      u32x4 g0 = {1u, lds, (unsigned)(ga & 0xFFFFFFFFull),
                  (unsigned)((ga >> 32) & 0x1FFFFFFull) | (2u << 30)};
      // D# group1: data_size=1 (2B), tile 32x16, row stride 512 elems
      unsigned long long dim0 = 1ull << 20, dim1 = 1ull << 20;
      unsigned long long stride0 = HDIM;
      unsigned long long q0 = (1ull << 16) | ((dim0 & 0xFFFFull) << 48);
      unsigned long long q1 = ((dim0 >> 16) & 0xFFFFull) |
                              ((dim1 & 0xFFFFFFFFull) << 16) | (32ull << 48);
      unsigned long long q2 = 16ull | ((stride0 & 0xFFFFFFFFull) << 32);
      unsigned long long q3 = (stride0 >> 32) & 0xFFFFull;
      i32x8 g1 = {(int)(unsigned)q0, (int)(unsigned)(q0 >> 32),
                  (int)(unsigned)q1, (int)(unsigned)(q1 >> 32),
                  (int)(unsigned)q2, (int)(unsigned)(q2 >> 32),
                  (int)(unsigned)q3, (int)(unsigned)(q3 >> 32)};
      i32x4 gz = {0, 0, 0, 0};
#if defined(__clang_major__) && __clang_major__ >= 23
      i32x8 gz8 = {0, 0, 0, 0, 0, 0, 0, 0};
      __builtin_amdgcn_tensor_load_to_lds(g0, g1, gz, gz, gz8, 0);
#else
      __builtin_amdgcn_tensor_load_to_lds(g0, g1, gz, gz, 0);
#endif
      __builtin_amdgcn_s_wait_tensorcnt(0);
    }
#else
    if (threadIdx.x < 64) {
      int r = threadIdx.x >> 2, q = threadIdx.x & 3;
      async_copy16(lds_addr(ldsA) + threadIdx.x * 16u,
                   &h1[(size_t)(row0 + r) * HDIM + k0 + q * 8]);
    }
    async_wait0();
#endif
    __syncthreads();
    v16bf a;
#pragma unroll
    for (int e = 0; e < 16; ++e)
      a[e] = ldsA[(lane & 15) * 32 + e + 16 * half];
    v16bf b = We2swz[(kc * 32 + nt) * 32 + lane];
    acc = wmma_bf16(a, b, acc);
  }
#pragma unroll
  for (int v = 0; v < 8; ++v) {
    int row = row0 + v + 8 * half;
    int col = col0 + nn;
    hout[(size_t)row * HDIM + col] = (bf16_t)(acc[v] + be2[col]);
  }
}

// -------- per-batch h.h^T tiles + fused sigmoid/target/BCE/mask/reduce
__global__ __launch_bounds__(32) void bce_kernel(
    const bf16_t* __restrict__ h, const int* __restrict__ Zpre,
    const float* __restrict__ Lpre, const int* __restrict__ ilens,
    float* __restrict__ accE) {
  int tj = blockIdx.x, ti = blockIdx.y, b = blockIdx.z;
  if (ti < tj) return;  // strictly upper triangle: target == 0 everywhere
  int lane = threadIdx.x & 31, half = lane >> 4, nn = lane & 15;
  int t0 = ti * 16, s0 = tj * 16;
  const bf16_t* hb = h + (size_t)b * TSZ * HDIM;
  v8f acc = {};
  for (int kc = 0; kc < 16; ++kc) {
    int k0 = kc * 32;
    v16bf a, bfr;
#pragma unroll
    for (int e = 0; e < 16; ++e) {
      a[e] = hb[(size_t)(t0 + (lane & 15)) * HDIM + k0 + e + 16 * half];
      bfr[e] = hb[(size_t)(s0 + nn) * HDIM + k0 + e + 16 * half];
    }
    acc = wmma_bf16(a, bfr, acc);
  }
  int len = ilens[b];
  const int* Zb = Zpre + b * TSZ;
  const float* Lb = Lpre + b * TSZ;
  float lsum = 0.0f;
#pragma unroll
  for (int v = 0; v < 8; ++v) {
    int t = t0 + v + 8 * half;
    int s = s0 + nn;
    if (t < s) continue;
    float tgt;
    if (t == s) {
      tgt = 1.0f;
    } else {
      int z = Zb[t - 1] - (s > 0 ? Zb[s - 1] : 0);
      tgt = (z > 0) ? 0.0f : expf(Lb[t - 1] - (s > 0 ? Lb[s - 1] : 0.0f));
    }
    bool valid = (t < len) && (s < len);
    if (valid && tgt > 0.1f) {
      float kv = 1.0f / (1.0f + expf(-acc[v]));
      float lk = fmaxf(logf(kv), -100.0f);
      float l1 = fmaxf(log1pf(-kv), -100.0f);
      lsum += -(tgt * lk + (1.0f - tgt) * l1);
    }
  }
  lsum = wave_reduce(lsum);
  if (lane == 0) atomicAdd(accE, lsum);
}

// ---------------------------------------------------------------- host
extern "C" void kernel_launch(void* const* d_in, const int* in_sizes, int n_in,
                              void* d_out, int out_size, void* d_ws,
                              size_t ws_size, hipStream_t stream) {
  const float* x = (const float*)d_in[0];
  const float* xo = (const float*)d_in[1];
  const int* il = (const int*)d_in[2];
  const float* Wt1 = (const float*)d_in[3];
  const float* bt1 = (const float*)d_in[4];
  const float* Wt2 = (const float*)d_in[5];
  const float* bt2 = (const float*)d_in[6];
  const float* We1 = (const float*)d_in[7];
  const float* be1 = (const float*)d_in[8];
  const float* We2 = (const float*)d_in[9];
  const float* be2 = (const float*)d_in[10];
  float* out = (float*)d_out;

  char* ws = (char*)d_ws;
  size_t off = 0;
  auto alloc = [&](size_t bytes) -> void* {
    off = (off + 255) & ~(size_t)255;
    void* p = ws + off;
    off += bytes;
    return p;
  };
  // fragment-major weights (KT*NT*32 lane-vectors of 16 bf16)
  bf16_t* Wt1swzB = (bf16_t*)alloc((size_t)3 * 32 * 32 * 16 * 2);
  bf16_t* Wt2swzB = (bf16_t*)alloc((size_t)16 * 6 * 32 * 16 * 2);
  bf16_t* We1swzB = (bf16_t*)alloc((size_t)3 * 32 * 32 * 16 * 2);
  bf16_t* We2swzB = (bf16_t*)alloc((size_t)16 * 32 * 32 * 16 * 2);
  bf16_t* Wt1swzA = (bf16_t*)alloc((size_t)16 * 5 * 32 * 16 * 2);
  bf16_t* xpad = (bf16_t*)alloc((size_t)NROW * IPAD * 2);
  bf16_t* g1 = (bf16_t*)alloc((size_t)NROW * HDIM * 2);
  bf16_t* g2 = (bf16_t*)alloc((size_t)NROW * HDIM * 2);
  unsigned short* rm16 = (unsigned short*)alloc((size_t)NROW * 32 * 2);
  signed char* esign = (signed char*)alloc((size_t)NROW * IDIM);
  bf16_t* h1 = (bf16_t*)alloc((size_t)NROW * HDIM * 2);
  bf16_t* hbuf = (bf16_t*)alloc((size_t)NROW * HDIM * 2);
  float* eraw = (float*)alloc((size_t)NROW * 4);
  float* mm = (float*)alloc(2 * 4);
  int* Zpre = (int*)alloc((size_t)BSZ * TSZ * 4);
  float* Lpre = (float*)alloc((size_t)BSZ * TSZ * 4);
  float* accs = (float*)alloc(2 * 4);

  init_kernel<<<1, 32, 0, stream>>>(accs);
  swizzle_b_kernel<<<(3 * 32 * 32 + 255) / 256, 256, 0, stream>>>(
      Wt1, Wt1swzB, IDIM, HDIM, 3, 32);
  swizzle_b_kernel<<<(16 * 6 * 32 + 255) / 256, 256, 0, stream>>>(
      Wt2, Wt2swzB, HDIM, IDIM, 16, 6);
  swizzle_b_kernel<<<(3 * 32 * 32 + 255) / 256, 256, 0, stream>>>(
      We1, We1swzB, IDIM, HDIM, 3, 32);
  swizzle_b_kernel<<<(16 * 32 * 32 + 255) / 256, 256, 0, stream>>>(
      We2, We2swzB, HDIM, HDIM, 16, 32);
  swizzle_a_kernel<<<(16 * 5 * 32 + 255) / 256, 256, 0, stream>>>(
      Wt1, Wt1swzA, IDIM, HDIM, 16, 5);
  int n4 = NROW * IPAD;
  cast_pad_kernel<<<(n4 + 255) / 256, 256, 0, stream>>>(x, xpad, NROW, IDIM,
                                                        NROW, IPAD);

  dim3 gIn(HDIM / 64, NROW / 16);
  mlp_in_kernel<<<gIn, 128, 0, stream>>>(xpad, (const v16bf*)Wt1swzB, bt1, g1,
                                         g2, rm16, 1);
  mlp_t2_kernel<<<NROW / 16, 192, 0, stream>>>(g1, g2, (const v16bf*)Wt2swzB,
                                               bt2, x, xo, il, esign, accs);
  energy_kernel<<<NROW, 160, 0, stream>>>((const v16bf*)Wt1swzA,
                                          (const v16bf*)Wt2swzB, rm16, esign, x,
                                          eraw);
  minmax_kernel<<<1, 256, 0, stream>>>(eraw, mm);
  rank_kernel<<<BSZ, TSZ, 0, stream>>>(eraw, mm, Zpre, Lpre);
  mlp_in_kernel<<<gIn, 128, 0, stream>>>(xpad, (const v16bf*)We1swzB, be1, h1,
                                         nullptr, nullptr, 0);
  mlp_h2_kernel<<<dim3(HDIM / 64, NROW / 16), 128, 0, stream>>>(
      h1, (const v16bf*)We2swzB, be2, hbuf);
  bce_kernel<<<dim3(TSZ / 16, TSZ / 16, BSZ), 32, 0, stream>>>(hbuf, Zpre, Lpre,
                                                               il, accs + 1);
  combine_kernel<<<1, 1, 0, stream>>>(accs, out);
}
